// GroupedQueryAttention_84370337563208
// MI455X (gfx1250) — matrix operand order repaired and checked by reading, not compiled, not run on hardware
//
#include <hip/hip_runtime.h>
#include <stdint.h>

#define Bq 4
#define Sq 1024
#define DIM 2048
#define NH 16
#define NKV 4
#define HD 128

typedef __attribute__((ext_vector_type(16))) __bf16 v16bf;
typedef __attribute__((ext_vector_type(8)))  float  v8f;

#if __has_builtin(__builtin_amdgcn_global_load_async_to_lds_b128) && \
    __has_builtin(__builtin_amdgcn_s_wait_asynccnt)
#define HAVE_ASYNC_LDS 1
typedef int v4i_vec __attribute__((vector_size(4 * sizeof(int))));
typedef __attribute__((address_space(1))) v4i_vec GlobalV4i;  // global src
typedef __attribute__((address_space(3))) v4i_vec LdsV4i;     // LDS dst
#endif

__device__ __forceinline__ v8f wmma_bf16(v16bf a, v16bf b, v8f c) {
  return __builtin_amdgcn_wmma_f32_16x16x32_bf16(false, a, false, b, (short)0, c, false, false);
}

// ---- Fragment loaders (wave32, v_wmma_f32_16x16x32_bf16) ----
// A-matrix 16x32 (MxK) from row-major source: lane = M (lane&15);
// lanes 0-15 hold K 0..15, lanes 16-31 hold K 16..31 (contiguous per lane).
template <typename T>
__device__ __forceinline__ v16bf frag_a(const T* base, int ld) {
  const int lane = threadIdx.x & 31;
  const T* p = base + (size_t)(lane & 15) * ld + ((lane >> 4) << 4);
  v16bf f;
#pragma unroll
  for (int i = 0; i < 4; ++i) {
    f[2 * i]         = (__bf16)p[2 * i];
    f[2 * i + 1]     = (__bf16)p[2 * i + 1];
    f[8 + 2 * i]     = (__bf16)p[8 + 2 * i];
    f[8 + 2 * i + 1] = (__bf16)p[8 + 2 * i + 1];
  }
  return f;
}

// B-matrix 32x16 (KxN), source row-major N x K (B = src^T): lane = N;
// lanes 0-15 hold K 0..15, lanes 16-31 hold K 16..31 (contiguous per lane).
template <typename T>
__device__ __forceinline__ v16bf frag_b_row(const T* base, int ld) {
  const int lane = threadIdx.x & 31;
  const T* p = base + (size_t)(lane & 15) * ld + ((lane >> 4) << 4);
  v16bf f;
#pragma unroll
  for (int i = 0; i < 16; ++i) f[i] = (__bf16)p[i];
  return f;
}

// ---- GEMM: C[M,N_] = A[M,K_] * W[N_,K_]^T, compile-time shapes ----
template <typename TA, int N_, int K_>
__global__ __launch_bounds__(256) void gemm_xwt_kernel(
    const TA* __restrict__ A, const float* __restrict__ W,
    float* __restrict__ C) {
  __shared__ __align__(16) __bf16 sA[128 * 32];
  __shared__ __align__(16) __bf16 sW[128 * 32];
  const int tid  = threadIdx.x;
  const int lane = tid & 31;
  const int wave = tid >> 5;   // 0..7
  const int wm   = wave >> 1;  // 0..3 -> 32-row strip
  const int wn   = wave & 1;   // 0..1 -> 64-col strip
  const int m0 = blockIdx.y * 128;
  const int n0 = blockIdx.x * 128;

  v8f acc[2][4];
#pragma unroll
  for (int i = 0; i < 2; ++i)
#pragma unroll
    for (int j = 0; j < 4; ++j)
#pragma unroll
      for (int r = 0; r < 8; ++r) acc[i][j][r] = 0.0f;

  // staging: each thread moves 16 CONTIGUOUS elements (vectorizable)
  const int srow = tid >> 1;          // 0..127
  const int scol = (tid & 1) * 16;    // 0 or 16

  for (int k0 = 0; k0 < K_; k0 += 32) {
    const TA*    pa = A + (size_t)(m0 + srow) * K_ + k0 + scol;
    const float* pw = W + (size_t)(n0 + srow) * K_ + k0 + scol;
    if (k0 + 32 < K_) {
      __builtin_prefetch(pa + 32, 0, 1);
      __builtin_prefetch(pw + 32, 0, 1);
    }
#pragma unroll
    for (int i = 0; i < 16; ++i) {
      sA[srow * 32 + scol + i] = (__bf16)pa[i];
      sW[srow * 32 + scol + i] = (__bf16)pw[i];
    }
    __syncthreads();
    v16bf af[2], wf[4];
#pragma unroll
    for (int i = 0; i < 2; ++i) af[i] = frag_a(sA + (wm * 32 + i * 16) * 32, 32);
#pragma unroll
    for (int j = 0; j < 4; ++j) wf[j] = frag_b_row(sW + (wn * 64 + j * 16) * 32, 32);
#pragma unroll
    for (int i = 0; i < 2; ++i)
#pragma unroll
      for (int j = 0; j < 4; ++j) acc[i][j] = wmma_bf16(af[i], wf[j], acc[i][j]);
    __syncthreads();
  }

  const int n  = lane & 15;
  const int hf = lane >> 4;
  float* cbase = C + (size_t)(m0 + wm * 32 + hf * 8) * N_ + (n0 + wn * 64 + n);
#pragma unroll
  for (int i = 0; i < 2; ++i)
#pragma unroll
    for (int j = 0; j < 4; ++j)
#pragma unroll
      for (int r = 0; r < 8; ++r)
        cbase[(size_t)(i * 16 + r) * N_ + j * 16] = acc[i][j][r];
}

// ---- RoPE on Q: read f32 [B*S][NH*HD], write bf16 head-major [B][NH][S][HD]
__global__ void rope_q_kernel(const float* __restrict__ Qf,
                              const float* __restrict__ cosb,
                              const float* __restrict__ sinb,
                              __bf16* __restrict__ Qbf) {
  size_t idx = (size_t)blockIdx.x * blockDim.x + threadIdx.x;  // B*S*NH*64
  int p = (int)(idx & 63);
  size_t t2 = idx >> 6;
  int h = (int)(t2 & (NH - 1));
  size_t row = t2 >> 4;              // b*S + s
  int s = (int)(row & (Sq - 1));
  int b = (int)(row >> 10);
  float c  = cosb[s * 64 + p];
  float sn = sinb[s * 64 + p];
  const float* q = Qf + row * (NH * HD) + h * HD + 2 * p;
  float xr = q[0], xi = q[1];
  __bf16* o = Qbf + ((((size_t)b * NH + h) * Sq + s) * HD + 2 * p);
  o[0] = (__bf16)(xr * c - xi * sn);
  o[1] = (__bf16)(xr * sn + xi * c);
}

// ---- RoPE on K: in-place f32 cache + bf16 head-major copy [B][NKV][S][HD]
__global__ void rope_k_kernel(float* __restrict__ Kc,
                              const float* __restrict__ cosb,
                              const float* __restrict__ sinb,
                              __bf16* __restrict__ Kbf) {
  size_t idx = (size_t)blockIdx.x * blockDim.x + threadIdx.x;  // B*S*NKV*64
  int p = (int)(idx & 63);
  size_t t2 = idx >> 6;
  int kvh = (int)(t2 & (NKV - 1));
  size_t row = t2 >> 2;
  int s = (int)(row & (Sq - 1));
  int b = (int)(row >> 10);
  float c  = cosb[s * 64 + p];
  float sn = sinb[s * 64 + p];
  float* k = Kc + row * (NKV * HD) + kvh * HD + 2 * p;
  float xr = k[0], xi = k[1];
  float orr = xr * c - xi * sn;
  float oii = xr * sn + xi * c;
  k[0] = orr;
  k[1] = oii;
  __bf16* o = Kbf + ((((size_t)b * NKV + kvh) * Sq + s) * HD + 2 * p);
  o[0] = (__bf16)orr;
  o[1] = (__bf16)oii;
}

// ---- V: f32 [B][S][NKV][HD] -> bf16 TRANSPOSED per head [B][NKV][HD][S]
__global__ void conv_v_kernel(const float* __restrict__ Vc,
                              __bf16* __restrict__ Vt) {
  size_t idx = (size_t)blockIdx.x * blockDim.x + threadIdx.x;  // B*S*NKV*HD
  int d = (int)(idx & (HD - 1));
  size_t t2 = idx >> 7;
  int kvh = (int)(t2 & (NKV - 1));
  size_t row = t2 >> 2;              // b*S + s
  int s = (int)(row & (Sq - 1));
  int b = (int)(row >> 10);
  Vt[(((size_t)b * NKV + kvh) * HD + d) * Sq + s] = (__bf16)Vc[idx];
}

// ---- Flash attention: 4 waves/block; block = 64 q rows of one (b,h). ----
// Qbf [B][NH][S][HD], Kbf [B][NKV][S][HD], Vt [B][NKV][HD][S], O bf16 [B*S][NH*HD]
__global__ __launch_bounds__(128) void attn_kernel(
    const __bf16* __restrict__ Qbf, const __bf16* __restrict__ Kbf,
    const __bf16* __restrict__ Vt, __bf16* __restrict__ O) {
  __shared__ __align__(16) __bf16 sK[32 * HD];       // 8KB  (32 keys x 128)
  __shared__ __align__(16) __bf16 sVt[HD * 32];      // 8KB  (128 hd x 32 keys)
  __shared__ __align__(16) __bf16 sP[4][16 * 32];    // 4KB  per-wave P tiles
  const int tid  = threadIdx.x;
  const int lane = tid & 31;
  const int wv   = tid >> 5;
  const int n  = lane & 15;
  const int hf = lane >> 4;

  int bid = blockIdx.x;              // B*NH*(S/64)
  int qb = bid & 15;
  int t  = bid >> 4;
  int h  = t & (NH - 1);
  int b  = t >> 4;
  int kvh = h >> 2;                  // G = 4
  const int Q0 = qb * 64;
  const int q0 = Q0 + wv * 16;

  const __bf16* qhead = Qbf + (((size_t)b * NH + h) * Sq) * HD;
  const __bf16* khead = Kbf + (((size_t)b * NKV + kvh) * Sq) * HD;
  const __bf16* vhead = Vt  + (((size_t)b * NKV + kvh) * HD) * Sq;

  v16bf qf[4];
#pragma unroll
  for (int c = 0; c < 4; ++c)
    qf[c] = frag_a(qhead + (size_t)q0 * HD + c * 32, HD);

  v8f o[8];
#pragma unroll
  for (int c = 0; c < 8; ++c)
#pragma unroll
    for (int r = 0; r < 8; ++r) o[c][r] = 0.0f;
  float m_r[8], l_r[8];
#pragma unroll
  for (int r = 0; r < 8; ++r) { m_r[r] = -1e30f; l_r[r] = 0.0f; }

  const float scale = 0.08838834764831845f;  // 1/sqrt(128)

  for (int kb = 0; kb < Q0 + 64; kb += 32) {
    // ---- cooperative tile staging: K (32x128) and V^T (128x32) ----
#if defined(HAVE_ASYNC_LDS)
    {
      const __bf16* gk = khead + (size_t)kb * HD;  // 4096 contiguous bf16
#pragma unroll
      for (int i = 0; i < 4; ++i) {
        unsigned e = (unsigned)tid * 8 + i * 1024;  // 8 bf16 = 16B per lane
        __builtin_amdgcn_global_load_async_to_lds_b128(
            (GlobalV4i*)(gk + e), (LdsV4i*)(sK + e), 0, 0);
        unsigned hd = e >> 5, kk = e & 31;
        __builtin_amdgcn_global_load_async_to_lds_b128(
            (GlobalV4i*)(vhead + (size_t)hd * Sq + kb + kk),
            (LdsV4i*)(sVt + e), 0, 0);
      }
      __builtin_amdgcn_s_wait_asynccnt(0);
    }
#else
    {
      const uint4* g4 = (const uint4*)(khead + (size_t)kb * HD);
      uint4* s4 = (uint4*)sK;
#pragma unroll
      for (int i = 0; i < 4; ++i) s4[tid + i * 128] = g4[tid + i * 128];
#pragma unroll
      for (int i = 0; i < 4; ++i) {
        unsigned e = (unsigned)tid * 8 + i * 1024;
        unsigned hd = e >> 5, kk = e & 31;
        *(uint4*)(sVt + e) = *(const uint4*)(vhead + (size_t)hd * Sq + kb + kk);
      }
    }
#endif
    __syncthreads();

    if (kb < q0 + 16) {  // wave-uniform causal skip
      // scores: 16x32 = two C frags, contract over HD (4 chunks of 32)
      v8f s0, s1;
#pragma unroll
      for (int r = 0; r < 8; ++r) { s0[r] = 0.0f; s1[r] = 0.0f; }
#pragma unroll
      for (int c = 0; c < 4; ++c) {
        v16bf k0f = frag_b_row(sK + c * 32, HD);
        s0 = wmma_bf16(qf[c], k0f, s0);
        v16bf k1f = frag_b_row(sK + 16 * HD + c * 32, HD);
        s1 = wmma_bf16(qf[c], k1f, s1);
      }

      // scale + causal mask + online softmax
#pragma unroll
      for (int r = 0; r < 8; ++r) {
        int qrow = q0 + hf * 8 + r;
        s0[r] = s0[r] * scale + ((kb + n)      <= qrow ? 0.0f : -1e9f);
        s1[r] = s1[r] * scale + ((kb + 16 + n) <= qrow ? 0.0f : -1e9f);
        float mx = fmaxf(s0[r], s1[r]);
#pragma unroll
        for (int off = 8; off > 0; off >>= 1) mx = fmaxf(mx, __shfl_xor(mx, off, 32));
        float mnew = fmaxf(m_r[r], mx);
        float p0 = __expf(s0[r] - mnew);
        float p1 = __expf(s1[r] - mnew);
        s0[r] = p0; s1[r] = p1;
        float ps = p0 + p1;
#pragma unroll
        for (int off = 8; off > 0; off >>= 1) ps += __shfl_xor(ps, off, 32);
        float corr = __expf(m_r[r] - mnew);
        l_r[r] = l_r[r] * corr + ps;
        m_r[r] = mnew;
#pragma unroll
        for (int c = 0; c < 8; ++c) o[c][r] *= corr;
      }

      // P: C-frag layout -> per-wave LDS tile -> A-frag (same-wave DS is in-order)
#pragma unroll
      for (int r = 0; r < 8; ++r) {
        sP[wv][(hf * 8 + r) * 32 + n]      = (__bf16)s0[r];
        sP[wv][(hf * 8 + r) * 32 + 16 + n] = (__bf16)s1[r];
      }
      v16bf pf = frag_a((const __bf16*)sP[wv], 32);

      // O += P @ V : B frag from V^T tile (contiguous keys per lane)
#pragma unroll
      for (int c = 0; c < 8; ++c) {
        v16bf vf = frag_b_row(sVt + (c * 16) * 32, 32);
        o[c] = wmma_bf16(pf, vf, o[c]);
      }
    }
    __syncthreads();
  }

  // normalize + store bf16 O [B*S][NH*HD]
#pragma unroll
  for (int r = 0; r < 8; ++r) {
    float inv = 1.0f / l_r[r];
    size_t row = (size_t)b * Sq + q0 + hf * 8 + r;
#pragma unroll
    for (int c = 0; c < 8; ++c)
      O[row * (NH * HD) + h * HD + c * 16 + n] = (__bf16)(o[c][r] * inv);
  }
}

extern "C" void kernel_launch(void* const* d_in, const int* in_sizes, int n_in,
                              void* d_out, int out_size, void* d_ws, size_t ws_size,
                              hipStream_t stream) {
  (void)in_sizes; (void)n_in; (void)out_size; (void)ws_size;
  const float* x  = (const float*)d_in[0];
  const float* fc = (const float*)d_in[1];
  const float* fs = (const float*)d_in[2];
  // d_in[3] = attention_mask (causal, applied analytically)
  const float* wq = (const float*)d_in[4];
  const float* wk = (const float*)d_in[5];
  const float* wv = (const float*)d_in[6];
  const float* wo = (const float*)d_in[7];

  float* out    = (float*)d_out;
  float* kcache = out + (size_t)Bq * Sq * NH * HD;       // post-RoPE K (f32)
  float* vcache = kcache + (size_t)Bq * Sq * NKV * HD;   // V (f32)

  char* w = (char*)d_ws;
  float*  Qf  = (float*)w;                                // 33,554,432 B
  __bf16* Qbf = (__bf16*)(w + 33554432);                  // 16,777,216 B
  __bf16* Kbf = (__bf16*)(w + 50331648);                  //  4,194,304 B
  __bf16* Vt  = (__bf16*)(w + 54525952);                  //  4,194,304 B
  __bf16* Obf = (__bf16*)(w + 58720256);                  // 16,777,216 B

  // Projections (C = X * W^T), all shapes compile-time
  gemm_xwt_kernel<float, NH * HD, DIM><<<dim3(16, 32), 256, 0, stream>>>(x, wq, Qf);
  gemm_xwt_kernel<float, NKV * HD, DIM><<<dim3(4, 32), 256, 0, stream>>>(x, wk, kcache);
  gemm_xwt_kernel<float, NKV * HD, DIM><<<dim3(4, 32), 256, 0, stream>>>(x, wv, vcache);

  // RoPE + bf16 re-layouts
  rope_q_kernel<<<(Bq * Sq * NH * 64) / 256, 256, 0, stream>>>(Qf, fc, fs, Qbf);
  rope_k_kernel<<<(Bq * Sq * NKV * 64) / 256, 256, 0, stream>>>(kcache, fc, fs, Kbf);
  conv_v_kernel<<<(Bq * Sq * NKV * HD) / 256, 256, 0, stream>>>(vcache, Vt);

  // Flash attention: B*NH*(S/64) blocks of 4 waves
  attn_kernel<<<Bq * NH * (Sq / 64), 128, 0, stream>>>(Qbf, Kbf, Vt, Obf);

  // Output projection
  gemm_xwt_kernel<__bf16, DIM, NH * HD><<<dim3(16, 32), 256, 0, stream>>>(Obf, wo, out);
}
